// iEFL_1623497638247
// MI455X (gfx1250) — matrix operand order repaired and checked by reading, not compiled, not run on hardware
//
#include <hip/hip_runtime.h>

// ---------------------------------------------------------------------------
// iEFL forward for MI455X (gfx1250, wave32).
// All dense GEMMs: v_wmma_f32_16x16x32_f16, f16 operands staged in workspace,
// 32x64 tile per wave (8 WMMA per k-step), all fragment loads are contiguous
// 16/32-byte vector loads.  Irregular parts (shape-mapper K=4, ROI-align,
// per-head softmax attention, LN, softmax-4096) are scalar VALU kernels.
// ---------------------------------------------------------------------------

typedef __attribute__((ext_vector_type(16))) _Float16 v16h;
typedef __attribute__((ext_vector_type(8)))  _Float16 v8h;
typedef __attribute__((ext_vector_type(8)))  float    v8f;

#define NB    16
#define CCH   256
#define WW    64
#define HWSZ  4096
#define KBOX  128
#define DIMM  256
#define ROWS  (NB * KBOX)        // 2048
#define NPTS  196                // (7*2)^2 ROI samples
#define ATTN_SCALE 0.17677669529663687f   // 32^-0.5
#define ROI_SCALE  0.125f

// ---------------- WMMA helpers (A/B k = 16*(lane/16) + s, contiguous; C/D per CDNA5 ISA 7.12.2) --------------

__device__ __forceinline__ v8f wmma16(v16h a, v16h b, v8f c) {
  return __builtin_amdgcn_wmma_f32_16x16x32_f16(
      /*neg_a=*/false, a, /*neg_b=*/false, b,
      /*c_mod=*/(short)0, c, /*reuse_a=*/false, /*reuse_b=*/false);
}

// A (16xK tile) from row-major f16, ld = row stride (multiple of 16).
// lane 0-15: row = mBase+lane, K = kb..kb+15
// lane 16-31: row = mBase+lane-16, K = kb+16..kb+31
__device__ __forceinline__ v16h load_a_h(const _Float16* __restrict__ X, int ld,
                                         int mBase, int kBase, int lane) {
  const _Float16* p = X + (size_t)(mBase + (lane & 15)) * ld + kBase + ((lane >> 4) << 4);
  v8h lo = *(const v8h*)p;          // 16B aligned
  v8h hi = *(const v8h*)(p + 8);
  return __builtin_shufflevector(lo, hi, 0, 1, 2, 3, 4, 5, 6, 7,
                                 8, 9, 10, 11, 12, 13, 14, 15);
}

// B (32x16 tile) where B[k,n] = F[n*ld + k]  (f16, per-lane contiguous 32B)
__device__ __forceinline__ v16h load_b_h(const _Float16* __restrict__ F, int ld,
                                         int nBase, int kBase, int lane) {
  return *(const v16h*)(F + (size_t)(nBase + (lane & 15)) * ld +
                        kBase + ((lane >> 4) << 4));   // 32B aligned
}

// store 16x16 f32 tile: lane 0-15 -> n = nBase+lane, rows mBase..mBase+7;
// lane 16-31 -> rows mBase+8..mBase+15.  Optional f32 and/or f16 destination.
__device__ __forceinline__ void store_tile(float* __restrict__ Y,
                                           _Float16* __restrict__ Y16, int ld,
                                           int mBase, int nBase, v8f acc,
                                           const float* __restrict__ bias,
                                           const float* __restrict__ res,
                                           int act, int lane) {
  int n  = nBase + (lane & 15);
  int m0 = mBase + ((lane >> 4) << 3);
  float bv = bias ? bias[n] : 0.0f;
#pragma unroll
  for (int r = 0; r < 8; ++r) {
    float v = acc[r] + bv;
    if (act) v = fmaxf(v, 0.0f);
    if (res) v += res[(size_t)(m0 + r) * ld + n];
    if (Y)   Y  [(size_t)(m0 + r) * ld + n] = v;
    if (Y16) Y16[(size_t)(m0 + r) * ld + n] = (_Float16)v;
  }
}

// ---------------- generic fused GEMM: Y = res + act(A*W^T + b) -------------
// A: (R x K) f16 row-major; W: (N x K) f16 row-major.  One wave per 32x64
// tile: 2 A-frags x 4 B-frags -> 8 WMMA per 32-wide k-step.
__global__ void iefl_gemm_h(const _Float16* __restrict__ A, const _Float16* __restrict__ W,
                            const float* __restrict__ bias, const float* __restrict__ res,
                            float* __restrict__ Y, _Float16* __restrict__ Y16,
                            int R, int N, int K, int act) {
  int gw   = blockIdx.x * (blockDim.x >> 5) + (threadIdx.x >> 5);
  int lane = threadIdx.x & 31;
  int ntiles = N >> 6;
  int tot    = (R >> 5) * ntiles;
  if (gw >= tot) return;                       // wave-uniform
  int mBase = (gw / ntiles) << 5;
  int nBase = (gw % ntiles) << 6;
  v8f c00 = {}, c01 = {}, c02 = {}, c03 = {};
  v8f c10 = {}, c11 = {}, c12 = {}, c13 = {};
  for (int kb = 0; kb < K; kb += 32) {
    v16h a0 = load_a_h(A, K, mBase,      kb, lane);
    v16h a1 = load_a_h(A, K, mBase + 16, kb, lane);
    v16h b0 = load_b_h(W, K, nBase +  0, kb, lane);
    v16h b1 = load_b_h(W, K, nBase + 16, kb, lane);
    v16h b2 = load_b_h(W, K, nBase + 32, kb, lane);
    v16h b3 = load_b_h(W, K, nBase + 48, kb, lane);
    c00 = wmma16(a0, b0, c00); c01 = wmma16(a0, b1, c01);
    c02 = wmma16(a0, b2, c02); c03 = wmma16(a0, b3, c03);
    c10 = wmma16(a1, b0, c10); c11 = wmma16(a1, b1, c11);
    c12 = wmma16(a1, b2, c12); c13 = wmma16(a1, b3, c13);
  }
  store_tile(Y, Y16, N, mBase,      nBase +  0, c00, bias, res, act, lane);
  store_tile(Y, Y16, N, mBase,      nBase + 16, c01, bias, res, act, lane);
  store_tile(Y, Y16, N, mBase,      nBase + 32, c02, bias, res, act, lane);
  store_tile(Y, Y16, N, mBase,      nBase + 48, c03, bias, res, act, lane);
  store_tile(Y, Y16, N, mBase + 16, nBase +  0, c10, bias, res, act, lane);
  store_tile(Y, Y16, N, mBase + 16, nBase + 16, c11, bias, res, act, lane);
  store_tile(Y, Y16, N, mBase + 16, nBase + 32, c12, bias, res, act, lane);
  store_tile(Y, Y16, N, mBase + 16, nBase + 48, c13, bias, res, act, lane);
}

// ---- linear attention logits: S[b] = Xla[b](128x256) * Fi[b](256x4096) ----
// B[k=c, n=m] = FiT[b][m*256 + c]  (transposed f16 copy -> contiguous loads)
__global__ void iefl_lattn_logits(const _Float16* __restrict__ XLA16,
                                  const _Float16* __restrict__ fiT16,
                                  float* __restrict__ S) {
  int gw   = blockIdx.x * (blockDim.x >> 5) + (threadIdx.x >> 5);
  int lane = threadIdx.x & 31;
  const int tilesPerB = 4 * 64;                // (128/32)*(4096/64)
  if (gw >= NB * tilesPerB) return;
  int b  = gw / tilesPerB;
  int t  = gw % tilesPerB;
  int mBase = b * KBOX + ((t >> 6) << 5);      // absolute row in (B*128) x *
  int nBase = (t & 63) << 6;
  const _Float16* FbT = fiT16 + (size_t)b * HWSZ * CCH;
  v8f c00 = {}, c01 = {}, c02 = {}, c03 = {};
  v8f c10 = {}, c11 = {}, c12 = {}, c13 = {};
  for (int kb = 0; kb < CCH; kb += 32) {
    v16h a0 = load_a_h(XLA16, DIMM, mBase,      kb, lane);
    v16h a1 = load_a_h(XLA16, DIMM, mBase + 16, kb, lane);
    v16h b0 = load_b_h(FbT, CCH, nBase +  0, kb, lane);
    v16h b1 = load_b_h(FbT, CCH, nBase + 16, kb, lane);
    v16h b2 = load_b_h(FbT, CCH, nBase + 32, kb, lane);
    v16h b3 = load_b_h(FbT, CCH, nBase + 48, kb, lane);
    c00 = wmma16(a0, b0, c00); c01 = wmma16(a0, b1, c01);
    c02 = wmma16(a0, b2, c02); c03 = wmma16(a0, b3, c03);
    c10 = wmma16(a1, b0, c10); c11 = wmma16(a1, b1, c11);
    c12 = wmma16(a1, b2, c12); c13 = wmma16(a1, b3, c13);
  }
  store_tile(S, nullptr, HWSZ, mBase,      nBase +  0, c00, nullptr, nullptr, 0, lane);
  store_tile(S, nullptr, HWSZ, mBase,      nBase + 16, c01, nullptr, nullptr, 0, lane);
  store_tile(S, nullptr, HWSZ, mBase,      nBase + 32, c02, nullptr, nullptr, 0, lane);
  store_tile(S, nullptr, HWSZ, mBase,      nBase + 48, c03, nullptr, nullptr, 0, lane);
  store_tile(S, nullptr, HWSZ, mBase + 16, nBase +  0, c10, nullptr, nullptr, 0, lane);
  store_tile(S, nullptr, HWSZ, mBase + 16, nBase + 16, c11, nullptr, nullptr, 0, lane);
  store_tile(S, nullptr, HWSZ, mBase + 16, nBase + 32, c12, nullptr, nullptr, 0, lane);
  store_tile(S, nullptr, HWSZ, mBase + 16, nBase + 48, c13, nullptr, nullptr, 0, lane);
}

// ---- Out[b] = res + P[b](128x4096) * Fi[b]^T -> (128x256) -----------------
// B[k=m, n=c] = Fi16[b][c*4096 + m]  (natural layout -> contiguous loads)
__global__ void iefl_lattn_out(const _Float16* __restrict__ P16,
                               const _Float16* __restrict__ fi16,
                               const float* __restrict__ res,
                               float* __restrict__ Y) {
  int gw   = blockIdx.x * (blockDim.x >> 5) + (threadIdx.x >> 5);
  int lane = threadIdx.x & 31;
  const int tilesPerB = 4 * 4;                 // (128/32)*(256/64)
  if (gw >= NB * tilesPerB) return;
  int b  = gw / tilesPerB;
  int t  = gw % tilesPerB;
  int mBase = b * KBOX + ((t >> 2) << 5);
  int nBase = (t & 3) << 6;
  const _Float16* Fb = fi16 + (size_t)b * CCH * HWSZ;
  v8f c00 = {}, c01 = {}, c02 = {}, c03 = {};
  v8f c10 = {}, c11 = {}, c12 = {}, c13 = {};
  for (int kb = 0; kb < HWSZ; kb += 32) {
    v16h a0 = load_a_h(P16, HWSZ, mBase,      kb, lane);
    v16h a1 = load_a_h(P16, HWSZ, mBase + 16, kb, lane);
    v16h b0 = load_b_h(Fb, HWSZ, nBase +  0, kb, lane);
    v16h b1 = load_b_h(Fb, HWSZ, nBase + 16, kb, lane);
    v16h b2 = load_b_h(Fb, HWSZ, nBase + 32, kb, lane);
    v16h b3 = load_b_h(Fb, HWSZ, nBase + 48, kb, lane);
    c00 = wmma16(a0, b0, c00); c01 = wmma16(a0, b1, c01);
    c02 = wmma16(a0, b2, c02); c03 = wmma16(a0, b3, c03);
    c10 = wmma16(a1, b0, c10); c11 = wmma16(a1, b1, c11);
    c12 = wmma16(a1, b2, c12); c13 = wmma16(a1, b3, c13);
  }
  store_tile(Y, nullptr, DIMM, mBase,      nBase +  0, c00, nullptr, res, 0, lane);
  store_tile(Y, nullptr, DIMM, mBase,      nBase + 16, c01, nullptr, res, 0, lane);
  store_tile(Y, nullptr, DIMM, mBase,      nBase + 32, c02, nullptr, res, 0, lane);
  store_tile(Y, nullptr, DIMM, mBase,      nBase + 48, c03, nullptr, res, 0, lane);
  store_tile(Y, nullptr, DIMM, mBase + 16, nBase +  0, c10, nullptr, res, 0, lane);
  store_tile(Y, nullptr, DIMM, mBase + 16, nBase + 16, c11, nullptr, res, 0, lane);
  store_tile(Y, nullptr, DIMM, mBase + 16, nBase + 32, c12, nullptr, res, 0, lane);
  store_tile(Y, nullptr, DIMM, mBase + 16, nBase + 48, c13, nullptr, res, 0, lane);
}

// ---------------- data staging kernels -------------------------------------

// Fi (B,C,HW) f32 -> fi16 same layout + fiT16 (B,HW,C), LDS-tiled transpose.
__global__ void iefl_fi_pack(const float* __restrict__ Fi,
                             _Float16* __restrict__ fi16,
                             _Float16* __restrict__ fiT16) {
  __shared__ float tile[32][33];
  int m0 = blockIdx.x << 5;        // HW base
  int c0 = blockIdx.y << 5;        // C base
  int b  = blockIdx.z;
  int tx = threadIdx.x;            // 0..31 (along HW)
  int ty = threadIdx.y;            // 0..7
  const float* src = Fi + (size_t)b * CCH * HWSZ;
#pragma unroll
  for (int i = 0; i < 32; i += 8) {
    int c = c0 + ty + i;
    float v = src[(size_t)c * HWSZ + m0 + tx];
    tile[ty + i][tx] = v;
    fi16[((size_t)b * CCH + c) * HWSZ + m0 + tx] = (_Float16)v;
  }
  __syncthreads();
#pragma unroll
  for (int i = 0; i < 32; i += 8) {
    int m = m0 + ty + i;
    fiT16[((size_t)b * HWSZ + m) * CCH + c0 + tx] = (_Float16)tile[tx][ty + i];
  }
}

__global__ void iefl_cvt_f16(const float* __restrict__ src, _Float16* __restrict__ dst,
                             int n) {
  int i = blockIdx.x * blockDim.x + threadIdx.x;
  if (i < n) dst[i] = (_Float16)src[i];
}

// ---------------- scalar kernels -------------------------------------------

// shape mapper: relu(lin 4->256), relu(lin 256->256), lin 256->256  (block = row)
__global__ void iefl_shape_mapper(const float* __restrict__ bboxes,
                                  const float* __restrict__ w1, const float* __restrict__ b1,
                                  const float* __restrict__ w2, const float* __restrict__ b2,
                                  const float* __restrict__ w3, const float* __restrict__ b3,
                                  float* __restrict__ out) {
  __shared__ float h1[DIMM];
  __shared__ float h2[DIMM];
  int row = blockIdx.x, t = threadIdx.x;
  const float* bb = bboxes + (size_t)row * 4;
  float acc = b1[t];
#pragma unroll
  for (int i = 0; i < 4; ++i) acc += w1[t * 4 + i] * bb[i];
  h1[t] = fmaxf(acc, 0.0f);
  __syncthreads();
  acc = b2[t];
  const float* wr = w2 + (size_t)t * DIMM;
  for (int i = 0; i < DIMM; ++i) acc += wr[i] * h1[i];
  h2[t] = fmaxf(acc, 0.0f);
  __syncthreads();
  acc = b3[t];
  wr = w3 + (size_t)t * DIMM;
  for (int i = 0; i < DIMM; ++i) acc += wr[i] * h2[i];
  out[(size_t)row * DIMM + t] = acc;
}

// ROI align (mean over 14x14 samples); block = (b,k), thread = channel
__global__ void iefl_roi_align(const float* __restrict__ Fi,
                               const float* __restrict__ bboxes,
                               float* __restrict__ roi,
                               _Float16* __restrict__ roi16) {
  __shared__ int   s_lo[2][14], s_hi[2][14];
  __shared__ float s_lf[2][14], s_vm[2][14];
  int row = blockIdx.x;                  // b*128 + k
  int b   = row >> 7;
  int t   = threadIdx.x;

  int axis = -1, j = 0;
  if (t < 14)                 { axis = 0; j = t; }        // y axis
  else if (t >= 16 && t < 30) { axis = 1; j = t - 16; }   // x axis
  if (axis >= 0) {
    const float* bb = bboxes + (size_t)row * 4;
    float x1 = bb[0] * ROI_SCALE, y1 = bb[1] * ROI_SCALE;
    float x2 = bb[2] * ROI_SCALE, y2 = bb[3] * ROI_SCALE;
    float binw = fmaxf(x2 - x1, 1.0f) * (1.0f / 7.0f);
    float binh = fmaxf(y2 - y1, 1.0f) * (1.0f / 7.0f);
    float g  = (float)(j >> 1) + ((float)(j & 1) + 0.5f) * 0.5f;
    float c  = (axis == 0) ? (y1 + g * binh) : (x1 + g * binw);
    float vm = (c > -1.0f && c < 64.0f) ? 1.0f : 0.0f;
    float cc = fmaxf(c, 0.0f);
    int lo = min((int)floorf(cc), 63);
    int hi = min(lo + 1, 63);
    if (lo >= 63) cc = (float)lo;
    s_lo[axis][j] = lo; s_hi[axis][j] = hi;
    s_lf[axis][j] = cc - (float)lo; s_vm[axis][j] = vm;
  }
  __syncthreads();

  const float* plane = Fi + ((size_t)b * CCH + t) * HWSZ;
  float acc = 0.0f;
  for (int py = 0; py < 14; ++py) {
    int yl = s_lo[0][py], yh = s_hi[0][py];
    float ly = s_lf[0][py], vmy = s_vm[0][py];
    float hy = 1.0f - ly;
    for (int px = 0; px < 14; ++px) {
      int xl = s_lo[1][px], xh = s_hi[1][px];
      float lx = s_lf[1][px];
      float hx = 1.0f - lx;
      float v1 = plane[yl * WW + xl];
      float v2 = plane[yl * WW + xh];
      float v3 = plane[yh * WW + xl];
      float v4 = plane[yh * WW + xh];
      acc += vmy * s_vm[1][px] * (hy * (hx * v1 + lx * v2) + ly * (hx * v3 + lx * v4));
    }
  }
  float r = acc * (1.0f / (float)NPTS);
  roi  [(size_t)row * CCH + t] = r;
  roi16[(size_t)row * CCH + t] = (_Float16)r;
}

// LayerNorm over last dim (256); block = row; writes f32 + f16
__global__ void iefl_layernorm(const float* __restrict__ X, const float* __restrict__ g,
                               const float* __restrict__ bta,
                               float* __restrict__ Y, _Float16* __restrict__ Y16) {
  __shared__ float red[DIMM];
  int row = blockIdx.x, t = threadIdx.x;
  float x = X[(size_t)row * DIMM + t];
  red[t] = x; __syncthreads();
  for (int st = 128; st > 0; st >>= 1) { if (t < st) red[t] += red[t + st]; __syncthreads(); }
  float mu = red[0] * (1.0f / DIMM);
  __syncthreads();
  float d = x - mu;
  red[t] = d * d; __syncthreads();
  for (int st = 128; st > 0; st >>= 1) { if (t < st) red[t] += red[t + st]; __syncthreads(); }
  float var = red[0] * (1.0f / DIMM);
  float v = d * rsqrtf(var + 1e-5f) * g[t] + bta[t];
  Y  [(size_t)row * DIMM + t] = v;
  Y16[(size_t)row * DIMM + t] = (_Float16)v;
}

// per-(b,h,n) MHCA: scores(128) -> softmax -> out(32); block = 128 threads
__global__ void iefl_mhca_attn(const float* __restrict__ Q, const float* __restrict__ KM,
                               const float* __restrict__ VM, _Float16* __restrict__ AO16) {
  __shared__ float prob[KBOX];
  __shared__ float red[KBOX];
  int blk = blockIdx.x;
  int n = blk & 127, bh = blk >> 7, h = bh & 7, b = bh >> 3;
  int m = threadIdx.x;
  const float* qrow = Q  + ((size_t)(b * KBOX + n) * DIMM) + h * 32;
  const float* krow = KM + ((size_t)(b * KBOX + m) * DIMM) + h * 32;
  float s = 0.0f;
#pragma unroll
  for (int d = 0; d < 32; ++d) s += qrow[d] * krow[d];
  s *= ATTN_SCALE;
  red[m] = s; __syncthreads();
  for (int st = 64; st > 0; st >>= 1) { if (m < st) red[m] = fmaxf(red[m], red[m + st]); __syncthreads(); }
  float mx = red[0];
  __syncthreads();
  float e = __expf(s - mx);
  prob[m] = e; red[m] = e; __syncthreads();
  for (int st = 64; st > 0; st >>= 1) { if (m < st) red[m] += red[m + st]; __syncthreads(); }
  float inv = 1.0f / red[0];
  __syncthreads();
  if (m < 32) {
    int d = m;
    float o = 0.0f;
    for (int mm = 0; mm < KBOX; ++mm)
      o += prob[mm] * VM[((size_t)(b * KBOX + mm) * DIMM) + h * 32 + d];
    AO16[((size_t)(b * KBOX + n) * DIMM) + h * 32 + d] = (_Float16)(o * inv);
  }
}

// softmax over rows of length 4096; reads f32 logits, writes f16 probs
__global__ void iefl_softmax4096(const float* __restrict__ S, _Float16* __restrict__ P16) {
  __shared__ float red[256];
  int row = blockIdx.x, t = threadIdx.x;
  size_t base = (size_t)row * HWSZ;
  float v[16];
  float mx = -3.0e38f;
#pragma unroll
  for (int i = 0; i < 16; ++i) { v[i] = S[base + i * 256 + t]; mx = fmaxf(mx, v[i]); }
  red[t] = mx; __syncthreads();
  for (int st = 128; st > 0; st >>= 1) { if (t < st) red[t] = fmaxf(red[t], red[t + st]); __syncthreads(); }
  mx = red[0];
  __syncthreads();
  float sum = 0.0f;
#pragma unroll
  for (int i = 0; i < 16; ++i) { v[i] = __expf(v[i] - mx); sum += v[i]; }
  red[t] = sum; __syncthreads();
  for (int st = 128; st > 0; st >>= 1) { if (t < st) red[t] += red[t + st]; __syncthreads(); }
  float inv = 1.0f / red[0];
#pragma unroll
  for (int i = 0; i < 16; ++i) P16[base + i * 256 + t] = (_Float16)(v[i] * inv);
}

// concat [x(256) | roi(256)] -> f16 512 per row (GEMM A operand)
__global__ void iefl_concat(const float* __restrict__ X, const float* __restrict__ R,
                            _Float16* __restrict__ C) {
  int idx = blockIdx.x * blockDim.x + threadIdx.x;
  if (idx >= ROWS * 512) return;
  int r = idx >> 9, j = idx & 511;
  float v = (j < DIMM) ? X[(size_t)r * DIMM + j] : R[(size_t)r * DIMM + (j - DIMM)];
  C[idx] = (_Float16)v;
}

// ---------------------------------------------------------------------------

struct IeflParams {
  const float *ffw, *ffb, *kw, *kb, *law, *lab;
  const float *ln1g, *ln1b, *ln2g, *ln2b, *ln3g, *ln3b;
  const float *ow, *ob, *qw, *qb;
  const float *sm1w, *sm1b, *sm2w, *sm2b, *sm3w, *sm3b, *vw, *vb;
};

extern "C" void kernel_launch(void* const* d_in, const int* in_sizes, int n_in,
                              void* d_out, int out_size, void* d_ws, size_t ws_size,
                              hipStream_t stream) {
  const float* Fi     = (const float*)d_in[0];
  const float* bboxes = (const float*)d_in[1];

  IeflParams p;
  const float** L = (const float**)(&p);
  if (in_sizes[2] == 1024) {
    // insertion order: sm1 sm2 sm3 q k v o la ff ln1 ln2 ln3
    p.sm1w=(const float*)d_in[2];  p.sm1b=(const float*)d_in[3];
    p.sm2w=(const float*)d_in[4];  p.sm2b=(const float*)d_in[5];
    p.sm3w=(const float*)d_in[6];  p.sm3b=(const float*)d_in[7];
    p.qw  =(const float*)d_in[8];  p.qb  =(const float*)d_in[9];
    p.kw  =(const float*)d_in[10]; p.kb  =(const float*)d_in[11];
    p.vw  =(const float*)d_in[12]; p.vb  =(const float*)d_in[13];
    p.ow  =(const float*)d_in[14]; p.ob  =(const float*)d_in[15];
    p.law =(const float*)d_in[16]; p.lab =(const float*)d_in[17];
    p.ffw =(const float*)d_in[18]; p.ffb =(const float*)d_in[19];
    p.ln1g=(const float*)d_in[20]; p.ln1b=(const float*)d_in[21];
    p.ln2g=(const float*)d_in[22]; p.ln2b=(const float*)d_in[23];
    p.ln3g=(const float*)d_in[24]; p.ln3b=(const float*)d_in[25];
  } else {
    // jax tree_flatten order (keys sorted): ff k la ln1 ln2 ln3 o q sm1 sm2 sm3 v
    for (int i = 0; i < 24; ++i) L[i] = (const float*)d_in[2 + i];
  }

  // -------- workspace layout --------
  char* base = (char*)d_ws;
  size_t off = 0;
  auto alloc = [&](size_t bytes) { char* r = base + off; off += (bytes + 255) & ~size_t(255); return r; };
  _Float16* fi16   = (_Float16*)alloc((size_t)NB * CCH * HWSZ * 2);   // 33.5 MB (C,HW)
  _Float16* fiT16  = (_Float16*)alloc((size_t)NB * HWSZ * CCH * 2);   // 33.5 MB (HW,C)
  float*    S      = (float*)   alloc((size_t)ROWS * HWSZ * 4);       // 33.5 MB logits
  _Float16* P16    = (_Float16*)alloc((size_t)ROWS * HWSZ * 2);       // 16.8 MB probs
  float*    roi    = (float*)   alloc((size_t)ROWS * DIMM * 4);
  _Float16* roi16  = (_Float16*)alloc((size_t)ROWS * DIMM * 2);
  float*    Fa     = (float*)   alloc((size_t)ROWS * DIMM * 4);
  float*    Fb     = (float*)   alloc((size_t)ROWS * DIMM * 4);
  float*    T      = (float*)   alloc((size_t)ROWS * DIMM * 4);
  _Float16* T16    = (_Float16*)alloc((size_t)ROWS * DIMM * 2);
  float*    Qb     = (float*)   alloc((size_t)ROWS * DIMM * 4);
  float*    KM     = (float*)   alloc((size_t)ROWS * DIMM * 4);
  float*    VMb    = (float*)   alloc((size_t)ROWS * DIMM * 4);
  _Float16* AO16   = (_Float16*)alloc((size_t)ROWS * DIMM * 2);
  _Float16* XLA16  = (_Float16*)alloc((size_t)ROWS * DIMM * 2);
  _Float16* CAT16  = (_Float16*)alloc((size_t)ROWS * 512 * 2);
  _Float16* wq16   = (_Float16*)alloc((size_t)DIMM * DIMM * 2);
  _Float16* wk16   = (_Float16*)alloc((size_t)DIMM * DIMM * 2);
  _Float16* wv16   = (_Float16*)alloc((size_t)DIMM * DIMM * 2);
  _Float16* wo16   = (_Float16*)alloc((size_t)DIMM * DIMM * 2);
  _Float16* wla16  = (_Float16*)alloc((size_t)DIMM * DIMM * 2);
  _Float16* wff16  = (_Float16*)alloc((size_t)DIMM * 512 * 2);
  (void)ws_size;

  const int GB = 256;                 // 8 wave32s per block
  const int gemm256_blocks  = (ROWS / 32) * (DIMM / 64) / 8;   // 32
  const int logits_blocks   = NB * 4 * 64 / 8;                 // 512
  const int lattnout_blocks = NB * 4 * 4 / 8;                  // 32

  // ---- stage f16 operands ----
  iefl_fi_pack<<<dim3(HWSZ / 32, CCH / 32, NB), dim3(32, 8), 0, stream>>>(Fi, fi16, fiT16);
  iefl_cvt_f16<<<(DIMM * DIMM + 255) / 256, 256, 0, stream>>>(p.qw,  wq16,  DIMM * DIMM);
  iefl_cvt_f16<<<(DIMM * DIMM + 255) / 256, 256, 0, stream>>>(p.kw,  wk16,  DIMM * DIMM);
  iefl_cvt_f16<<<(DIMM * DIMM + 255) / 256, 256, 0, stream>>>(p.vw,  wv16,  DIMM * DIMM);
  iefl_cvt_f16<<<(DIMM * DIMM + 255) / 256, 256, 0, stream>>>(p.ow,  wo16,  DIMM * DIMM);
  iefl_cvt_f16<<<(DIMM * DIMM + 255) / 256, 256, 0, stream>>>(p.law, wla16, DIMM * DIMM);
  iefl_cvt_f16<<<(DIMM * 512  + 255) / 256, 256, 0, stream>>>(p.ffw, wff16, DIMM * 512);

  // ---- precompute ----
  iefl_shape_mapper<<<ROWS, 256, 0, stream>>>(bboxes, p.sm1w, p.sm1b, p.sm2w, p.sm2b,
                                              p.sm3w, p.sm3b, Fa);
  iefl_roi_align<<<ROWS, 256, 0, stream>>>(Fi, bboxes, roi, roi16);
  iefl_gemm_h<<<gemm256_blocks, GB, 0, stream>>>(roi16, wk16, p.kb, nullptr, KM, nullptr,
                                                 ROWS, DIMM, DIMM, 0);
  iefl_gemm_h<<<gemm256_blocks, GB, 0, stream>>>(roi16, wv16, p.vb, nullptr, VMb, nullptr,
                                                 ROWS, DIMM, DIMM, 0);

  float* F  = Fa;
  float* Fn = Fb;
  const size_t SLICE = (size_t)ROWS * DIMM * sizeof(float);   // 2 MB per output

  for (int it = 0; it < 2; ++it) {
    // ---- MHCA ----
    iefl_layernorm<<<ROWS, 256, 0, stream>>>(F, p.ln1g, p.ln1b, T, T16);
    iefl_gemm_h<<<gemm256_blocks, GB, 0, stream>>>(T16, wq16, p.qb, nullptr, Qb, nullptr,
                                                   ROWS, DIMM, DIMM, 0);
    iefl_mhca_attn<<<NB * 8 * KBOX, 128, 0, stream>>>(Qb, KM, VMb, AO16);
    iefl_gemm_h<<<gemm256_blocks, GB, 0, stream>>>(AO16, wo16, p.ob, F, Fn, nullptr,
                                                   ROWS, DIMM, DIMM, 0);
    { float* t = F; F = Fn; Fn = t; }
    hipMemcpyAsync((char*)d_out + (size_t)it * SLICE, F, SLICE,
                   hipMemcpyDeviceToDevice, stream);

    // ---- linear attention ----
    iefl_layernorm<<<ROWS, 256, 0, stream>>>(F, p.ln2g, p.ln2b, T, T16);
    iefl_gemm_h<<<gemm256_blocks, GB, 0, stream>>>(T16, wla16, p.lab, nullptr,
                                                   nullptr, XLA16, ROWS, DIMM, DIMM, 0);
    iefl_lattn_logits<<<logits_blocks, GB, 0, stream>>>(XLA16, fiT16, S);
    iefl_softmax4096<<<ROWS, 256, 0, stream>>>(S, P16);
    iefl_lattn_out<<<lattnout_blocks, GB, 0, stream>>>(P16, fi16, F, Fn);
    { float* t = F; F = Fn; Fn = t; }

    // ---- fusion ----
    iefl_layernorm<<<ROWS, 256, 0, stream>>>(F, p.ln3g, p.ln3b, T, T16);
    iefl_concat<<<(ROWS * 512 + 255) / 256, 256, 0, stream>>>(T, roi, CAT16);
    iefl_gemm_h<<<gemm256_blocks, GB, 0, stream>>>(CAT16, wff16, p.ffb, F, Fn, nullptr,
                                                   ROWS, DIMM, 512, 1);
    { float* t = F; F = Fn; Fn = t; }
  }

  hipMemcpyAsync((char*)d_out + 2 * SLICE, F, SLICE, hipMemcpyDeviceToDevice, stream);
  (void)n_in; (void)out_size;
}
